// W8A8Linear_22153441312853
// MI455X (gfx1250) — 8-bit matrix operand order repaired and checked by reading, compiled, not run on hardware
//
#include <hip/hip_runtime.h>
#include <hip/hip_fp16.h>
#include <stdint.h>

typedef int          v8i __attribute__((ext_vector_type(8)));
typedef int          v4i __attribute__((ext_vector_type(4)));
typedef unsigned int v4u __attribute__((ext_vector_type(4)));

typedef __attribute__((address_space(1))) v4i* gv4i_p;
typedef __attribute__((address_space(3))) v4i* lv4i_p;

#define M_DIM  256
#define K_DIM  8192
#define N_DIM  28672
#define QMAXF  127.0f
#define SMIN   1e-5f

#define KSLAB  128
#define NSLABS (K_DIM / KSLAB)
#define BN     128

#if defined(__AMDGCN__) && __has_builtin(__builtin_amdgcn_global_load_async_to_lds_b128)
#define HAVE_ASYNC_LDS 1
#else
#define HAVE_ASYNC_LDS 0
#endif

__device__ __forceinline__ void wait_async_all() {
#if HAVE_ASYNC_LDS
#if __has_builtin(__builtin_amdgcn_s_wait_asynccnt)
  __builtin_amdgcn_s_wait_asynccnt(0);
#else
  asm volatile("s_wait_asynccnt 0" ::: "memory");
#endif
#endif
}

__global__ __launch_bounds__(256)
void w8a8_quant_act(const __half* __restrict__ x,
                    int8_t* __restrict__ qx,
                    float* __restrict__ act_scale) {
  const int row = blockIdx.x;
  const int t   = threadIdx.x;
  const __half2* xrow = (const __half2*)(x + (size_t)row * K_DIM);

  __half2 v[4][4];
  float lmax = 0.0f;
#pragma unroll
  for (int i = 0; i < 4; ++i) {
    const int c = t + i * 256;
#pragma unroll
    for (int j = 0; j < 4; ++j) v[i][j] = xrow[c * 4 + j];
#pragma unroll
    for (int j = 0; j < 4; ++j) {
      const float2 f = __half22float2(v[i][j]);
      lmax = fmaxf(lmax, fmaxf(fabsf(f.x), fabsf(f.y)));
    }
  }

  __shared__ float red[256];
  red[t] = lmax;
  __syncthreads();
#pragma unroll
  for (int off = 128; off > 0; off >>= 1) {
    if (t < off) red[t] = fmaxf(red[t], red[t + off]);
    __syncthreads();
  }
  const float mx  = fmaxf(red[0], SMIN);
  const float scl = mx / QMAXF;
  const float inv = QMAXF / mx;
  if (t == 0) act_scale[row] = scl;

  int2* qrow = (int2*)(qx + (size_t)row * K_DIM);
#pragma unroll
  for (int i = 0; i < 4; ++i) {
    const int c = t + i * 256;
    int b[8];
#pragma unroll
    for (int j = 0; j < 4; ++j) {
      const float2 f = __half22float2(v[i][j]);
      b[2 * j]     = (int)rintf(f.x * inv);
      b[2 * j + 1] = (int)rintf(f.y * inv);
    }
    int2 o;
    o.x = (b[0] & 255) | ((b[1] & 255) << 8) | ((b[2] & 255) << 16) | (b[3] << 24);
    o.y = (b[4] & 255) | ((b[5] & 255) << 8) | ((b[6] & 255) << 16) | (b[7] << 24);
    qrow[c] = o;
  }
}

__global__ __launch_bounds__(256)
void w8a8_wmma_gemm(const int8_t* __restrict__ qx,
                    const float* __restrict__ act_scale,
                    const int8_t* __restrict__ w,
                    const float* __restrict__ scale_ch,
                    const __half* __restrict__ bias,
                    float* __restrict__ out) {
  __shared__ int8_t Abuf[2][M_DIM * KSLAB];

  const int t    = threadIdx.x;
  const int lane = t & 31;
  const int wave = t >> 5;
  const int hi   = lane >> 4;
  const int lq   = lane & 15;
  const int n0   = blockIdx.x * BN + wave * 16 + lq;

  const v8i vzero = {0, 0, 0, 0, 0, 0, 0, 0};
  v8i acc[16];
#pragma unroll
  for (int i = 0; i < 16; ++i) acc[i] = vzero;

  auto stage = [&](int slab, int buf) {
    const int8_t* g = qx + (size_t)t * K_DIM + (size_t)slab * KSLAB;
    int8_t* l = &Abuf[buf][t * KSLAB];
#if HAVE_ASYNC_LDS
#pragma unroll
    for (int c = 0; c < 8; ++c) {
      __builtin_amdgcn_global_load_async_to_lds_b128(
          (gv4i_p)(uintptr_t)(g + c * 16),
          (lv4i_p)(uint32_t)(uintptr_t)(l + c * 16),
          0, 0);
    }
#else
#pragma unroll
    for (int c = 0; c < 8; ++c)
      *(v4u*)(l + c * 16) = *(const v4u*)(g + c * 16);
#endif
  };

  auto compute = [&](int slab, int buf) {
    const size_t wrow = (size_t)n0 * K_DIM + (size_t)slab * KSLAB;
#pragma unroll
    for (int kk = 0; kk < KSLAB; kk += 64) {
      const v4u* wp = (const v4u*)(w + wrow + kk + hi * 16);
      const v4u b0 = __builtin_nontemporal_load(wp);
      const v4u b1 = __builtin_nontemporal_load(wp + 2);
      v8i bf;
      bf[0] = (int)b0[0]; bf[1] = (int)b0[1]; bf[2] = (int)b0[2]; bf[3] = (int)b0[3];
      bf[4] = (int)b1[0]; bf[5] = (int)b1[1]; bf[6] = (int)b1[2]; bf[7] = (int)b1[3];
#pragma unroll
      for (int mt = 0; mt < 16; ++mt) {
        const int arow = mt * 16 + lq;
        const int* lp = (const int*)&Abuf[buf][arow * KSLAB + kk + hi * 16];
        v8i af;
        af[0] = lp[0];  af[1] = lp[1];
        af[2] = lp[2];  af[3] = lp[3];
        af[4] = lp[8];  af[5] = lp[9];
        af[6] = lp[10]; af[7] = lp[11];
        acc[mt] = __builtin_amdgcn_wmma_i32_16x16x64_iu8(
            true, af, true, bf, acc[mt], false, false);
      }
    }
  };

  stage(0, 0);
  wait_async_all();
  __syncthreads();

  for (int s = 0; s < NSLABS; ++s) {
    if (s + 1 < NSLABS) stage(s + 1, (s + 1) & 1);
    compute(s, s & 1);
    wait_async_all();
    __syncthreads();
  }

  const float scn = scale_ch[n0];
  const float bn  = __half2float(bias[n0]);
#pragma unroll
  for (int mt = 0; mt < 16; ++mt) {
#pragma unroll
    for (int j = 0; j < 8; ++j) {
      const int m = mt * 16 + hi * 8 + j;
      const float r = (float)acc[mt][j] * act_scale[m] * scn + bn;
      out[(size_t)m * N_DIM + n0] = r;
    }
  }
}

extern "C" void kernel_launch(void* const* d_in, const int* in_sizes, int n_in,
                              void* d_out, int out_size, void* d_ws, size_t ws_size,
                              hipStream_t stream) {
  (void)in_sizes; (void)n_in; (void)out_size; (void)ws_size;
  const __half* x        = (const __half*)d_in[0];
  const int8_t* weight   = (const int8_t*)d_in[1];
  const float*  scale_ch = (const float*)d_in[2];
  const __half* bias     = (const __half*)d_in[3];
  float*        out      = (float*)d_out;

  int8_t* qx        = (int8_t*)d_ws;
  float*  act_scale = (float*)((char*)d_ws + (size_t)M_DIM * K_DIM);

  w8a8_quant_act<<<M_DIM, 256, 0, stream>>>(x, qx, act_scale);
  w8a8_wmma_gemm<<<N_DIM / BN, 256, 0, stream>>>(qx, act_scale, weight,
                                                 scale_ch, bias, out);
}
